// GraphAttentionLayer_18210661335429
// MI455X (gfx1250) — matrix operand order repaired and checked by reading, not compiled, not run on hardware
//
#include <hip/hip_runtime.h>

typedef __attribute__((ext_vector_type(16))) _Float16 v16h;
typedef __attribute__((ext_vector_type(8)))  _Float16 v8h;
typedef __attribute__((ext_vector_type(8)))  float    v8f;
typedef __attribute__((ext_vector_type(4)))  unsigned v4u;
typedef __attribute__((ext_vector_type(8)))  int      v8i;
typedef __attribute__((ext_vector_type(4)))  int      v4i;

#define B_        4
#define N_        2048
#define IN_DIM_   128
#define H_        4
#define D_        64
#define HD_       256
#define NEG_SLOPE_ 0.2f
#define MASK_VAL_ (-1e9f)

#define ROWB_     80                 // LDS row stride: 64B data + 16B TDM pad
#define LDS_BUF_  (D_ * ROWB_)       // 5120B per staged 32x64 hT chunk

static __device__ __forceinline__ v8f wmma_f16(v16h a, v16h b, v8f c) {
  return __builtin_amdgcn_wmma_f32_16x16x32_f16(false, a, false, b, (short)0, c,
                                                false, false);
}

static __device__ __forceinline__ v16h pack16(float4 a, float4 b, float4 c, float4 d) {
  v16h v;
  v[0]  = (_Float16)a.x; v[1]  = (_Float16)a.y; v[2]  = (_Float16)a.z; v[3]  = (_Float16)a.w;
  v[4]  = (_Float16)b.x; v[5]  = (_Float16)b.y; v[6]  = (_Float16)b.z; v[7]  = (_Float16)b.w;
  v[8]  = (_Float16)c.x; v[9]  = (_Float16)c.y; v[10] = (_Float16)c.z; v[11] = (_Float16)c.w;
  v[12] = (_Float16)d.x; v[13] = (_Float16)d.y; v[14] = (_Float16)d.z; v[15] = (_Float16)d.w;
  return v;
}

// bit-masked leaky-relu score -> softmax probability (f16)
static __device__ __forceinline__ _Float16 pval(unsigned bit, float ev, float es,
                                                float mi, float il) {
  float s = es + ev;
  s = s > 0.f ? s : NEG_SLOPE_ * s;
  s = bit ? s : MASK_VAL_;
  return (_Float16)(__expf(s - mi) * il);
}

// Tensor Data Mover: 2D tile 32 (j, contiguous) x 64 (d rows), 2B elements,
// row stride 2048 elements, LDS padding 4 DWORDs every 16 DWORDs (80B rows).
// 6-arg toolchain form: (g0 v4u, g1 v8i, g2 v4i, g3 v4i, v8i, cpol).
static __device__ __forceinline__ void tdm_load_tile(unsigned lds_addr,
                                                     unsigned long long gaddr) {
  v4u g0 = { 1u,                                 // count=1 (valid descriptor)
             lds_addr,                           // LDS byte address
             (unsigned)gaddr,                    // global addr [31:0]
             (unsigned)(gaddr >> 32) | (2u << 30) };  // addr[56:32] | type=2
  v8i g1 = { (int)((1u << 16)      // data_size = 2 bytes
                 | (1u << 20)      // pad_enable
                 | (3u << 22)      // pad_interval: 16 DWORDs (64B)
                 | (3u << 25)),    // pad_amount: 4 DWORDs (16B)
             (int)(2048u << 16),   // tensor_dim0 = 2048
             (int)(64u << 16),     // tensor_dim1 = 64
             (int)(32u << 16),     // tile_dim0 = 32
             64,                   // tile_dim1 = 64, tile_dim2 = 0
             2048,                 // tensor_dim0_stride = 2048
             0, 0 };
  v4i z4 = { 0, 0, 0, 0 };
  v8i z8 = { 0, 0, 0, 0, 0, 0, 0, 0 };
  __builtin_amdgcn_tensor_load_to_lds(g0, g1, z4, z4, z8, 0);
}

// ---------------------------------------------------------------------------
// Kernel 1: h = x @ W^T via WMMA f16, stored transposed as hT[b][h][d][n] f16.
// ---------------------------------------------------------------------------
__global__ void __launch_bounds__(256)
k_linear_hT(const float* __restrict__ x, const float* __restrict__ W,
            _Float16* __restrict__ hT) {
  const int lane = threadIdx.x & 31;
  const int wid  = (blockIdx.x * blockDim.x + threadIdx.x) >> 5;  // 0..2047
  const int head = wid & 3;
  const int rt   = wid >> 2;
  const int g0   = rt << 4;
  const int m16  = lane & 15;
  const bool hi  = lane >= 16;
  const int koff8  = hi ? 8 : 0;
  const int koff16 = hi ? 16 : 0;

  v8f acc[4] = {};
  const float* xrow = x + (size_t)(g0 + m16) * IN_DIM_;

#pragma unroll
  for (int kc = 0; kc < 4; ++kc) {
    const float* xp = xrow + kc * 32;
    float4 f0 = *(const float4*)(xp + koff8);
    float4 f1 = *(const float4*)(xp + koff8 + 4);
    float4 f2 = *(const float4*)(xp + 16 + koff8);
    float4 f3 = *(const float4*)(xp + 16 + koff8 + 4);
    v16h A = pack16(f0, f1, f2, f3);

#pragma unroll
    for (int ct = 0; ct < 4; ++ct) {
      const int o = head * 64 + ct * 16 + m16;
      const float* wp = W + (size_t)o * IN_DIM_ + kc * 32 + koff8;
      float4 w0 = *(const float4*)(wp);
      float4 w1 = *(const float4*)(wp + 4);
      float4 w2 = *(const float4*)(wp + 16);
      float4 w3 = *(const float4*)(wp + 20);
      acc[ct] = wmma_f16(A, pack16(w0, w1, w2, w3), acc[ct]);
    }
  }

  const int b    = g0 >> 11;
  const int n0   = g0 & (N_ - 1);
  const int roff = hi ? 8 : 0;
#pragma unroll
  for (int ct = 0; ct < 4; ++ct) {
    const int d = ct * 16 + m16;
    _Float16* hp = hT + ((size_t)((b * H_ + head) * D_ + d)) * N_ + n0 + roff;
#pragma unroll
    for (int r = 0; r < 8; ++r) hp[r] = (_Float16)acc[ct][r];
  }
}

// ---------------------------------------------------------------------------
// Kernel 2: pack adjacency into bitmasks: bits[i][j/32] (1 bit per edge).
// ---------------------------------------------------------------------------
__global__ void __launch_bounds__(256)
k_pack_adj(const int* __restrict__ adj, unsigned* __restrict__ bits) {
  const int w = blockIdx.x * 256 + threadIdx.x;  // 0..131071
  const int* p = adj + (size_t)w * 32;
  unsigned m = 0;
#pragma unroll
  for (int q = 0; q < 8; ++q) {
    int4 v = *(const int4*)(p + q * 4);
    m |= (v.x > 0 ? 1u : 0u) << (q * 4 + 0);
    m |= (v.y > 0 ? 1u : 0u) << (q * 4 + 1);
    m |= (v.z > 0 ? 1u : 0u) << (q * 4 + 2);
    m |= (v.w > 0 ? 1u : 0u) << (q * 4 + 3);
  }
  bits[w] = m;
}

// ---------------------------------------------------------------------------
// Kernel 3: e_src[b,h,n] = <h[b,h,n,:], att_src[h,:]>, same for e_dst.
// ---------------------------------------------------------------------------
__global__ void __launch_bounds__(256)
k_edge(const _Float16* __restrict__ hT, const float* __restrict__ att_src,
       const float* __restrict__ att_dst, float* __restrict__ e_src,
       float* __restrict__ e_dst) {
  const int t  = blockIdx.x * 256 + threadIdx.x;  // == bh*N + n
  const int n  = t & (N_ - 1);
  const int bh = t >> 11;
  const int h  = bh & 3;
  const _Float16* hp = hT + (size_t)bh * D_ * N_ + n;
  const float* as = att_src + h * D_;
  const float* ad = att_dst + h * D_;
  float ss = 0.f, sd = 0.f;
#pragma unroll 8
  for (int d = 0; d < D_; ++d) {
    float v = (float)hp[(size_t)d * N_];
    ss += v * as[d];
    sd += v * ad[d];
  }
  e_src[t] = ss;
  e_dst[t] = sd;
}

// ---------------------------------------------------------------------------
// Kernel 4: per-row softmax stats (max, 1/sum); scores recomputed (rank-1).
// ---------------------------------------------------------------------------
__global__ void __launch_bounds__(256)
k_stats(const unsigned* __restrict__ adjbits, const float* __restrict__ e_src,
        const float* __restrict__ e_dst, float* __restrict__ mrow,
        float* __restrict__ ilrow) {
  const int lane = threadIdx.x & 31;
  const int row  = (blockIdx.x * blockDim.x + threadIdx.x) >> 5;  // 0..32767
  const int i    = row & (N_ - 1);
  const int bh   = row >> 11;
  const float es = e_src[row];
  const unsigned* brow = adjbits + (size_t)i * (N_ / 32);
  const float* ed = e_dst + bh * N_;

  float mx = -3.4e38f;
  for (int t = 0; t < N_ / 32; ++t) {
    float s = es + ed[t * 32 + lane];
    s = s > 0.f ? s : NEG_SLOPE_ * s;
    s = ((brow[t] >> lane) & 1u) ? s : MASK_VAL_;
    mx = fmaxf(mx, s);
  }
#pragma unroll
  for (int off = 16; off >= 1; off >>= 1) mx = fmaxf(mx, __shfl_xor(mx, off, 32));

  float sum = 0.f;
  for (int t = 0; t < N_ / 32; ++t) {
    float s = es + ed[t * 32 + lane];
    s = s > 0.f ? s : NEG_SLOPE_ * s;
    s = ((brow[t] >> lane) & 1u) ? s : MASK_VAL_;
    sum += __expf(s - mx);
  }
#pragma unroll
  for (int off = 16; off >= 1; off >>= 1) sum += __shfl_xor(sum, off, 32);

  if (lane == 0) {
    mrow[row]  = mx;
    ilrow[row] = 1.0f / sum;
  }
}

// ---------------------------------------------------------------------------
// Kernel 5: out = softmax(scores) @ h, fused. Block = 8 waves sharing one
// (b,h); hT j-chunks staged to LDS by the Tensor Data Mover, double-buffered;
// probabilities built in registers as the WMMA A-matrix.
// ---------------------------------------------------------------------------
__global__ void __launch_bounds__(256)
k_attn(const unsigned* __restrict__ adjbits, const _Float16* __restrict__ hT,
       const float* __restrict__ e_src, const float* __restrict__ e_dst,
       const float* __restrict__ mrow, const float* __restrict__ ilrow,
       float* __restrict__ out) {
  __shared__ __align__(16) char smem[2 * LDS_BUF_];

  const int lane = threadIdx.x & 31;
  const int wav  = threadIdx.x >> 5;        // 0..7
  const int bh   = blockIdx.x >> 4;         // 0..15  (shared by whole block)
  const int it   = (blockIdx.x & 15) * 8 + wav;  // 0..127
  const int h    = bh & 3;
  const int b    = bh >> 2;
  const int i0   = it << 4;
  const int m16  = lane & 15;
  const bool hi  = lane >= 16;
  const int koff = hi ? 8 : 0;              // A-matrix K offset (upper half-wave)
  const int jbo  = hi ? 8 : 0;              // B-matrix K offset (upper half-wave)

  const int i    = i0 + m16;
  const int ridx = bh * N_ + i;
  const float es = e_src[ridx];
  const float mi = mrow[ridx];
  const float il = ilrow[ridx];

  const unsigned* brow = adjbits + (size_t)i * (N_ / 32);
  const float* ed = e_dst + bh * N_;

  const unsigned lds_base = (unsigned)(unsigned long long)(&smem[0]);
  const unsigned long long gbase =
      (unsigned long long)(&hT[(size_t)bh * D_ * N_]);

  v8f acc[4] = {};

  if (wav == 0) {                            // preload chunk 0
    tdm_load_tile(lds_base, gbase);
    __builtin_amdgcn_s_wait_tensorcnt(0);
  }
  __syncthreads();

  for (int c = 0; c < N_ / 32; ++c) {
    const int cur = c & 1;
    if ((wav == 0) && (c + 1 < N_ / 32))     // async-stage next chunk
      tdm_load_tile(lds_base + (unsigned)((cur ^ 1) * LDS_BUF_),
                    gbase + (unsigned long long)(c + 1) * 64ull);

    const int j0 = c * 32;
    const unsigned wv = brow[c];
    float4 e0 = *(const float4*)(ed + j0 + koff);
    float4 e1 = *(const float4*)(ed + j0 + koff + 4);
    float4 e2 = *(const float4*)(ed + j0 + 16 + koff);
    float4 e3 = *(const float4*)(ed + j0 + 16 + koff + 4);

    v16h A;
    A[0]  = pval((wv >> (koff + 0)) & 1u, e0.x, es, mi, il);
    A[1]  = pval((wv >> (koff + 1)) & 1u, e0.y, es, mi, il);
    A[2]  = pval((wv >> (koff + 2)) & 1u, e0.z, es, mi, il);
    A[3]  = pval((wv >> (koff + 3)) & 1u, e0.w, es, mi, il);
    A[4]  = pval((wv >> (koff + 4)) & 1u, e1.x, es, mi, il);
    A[5]  = pval((wv >> (koff + 5)) & 1u, e1.y, es, mi, il);
    A[6]  = pval((wv >> (koff + 6)) & 1u, e1.z, es, mi, il);
    A[7]  = pval((wv >> (koff + 7)) & 1u, e1.w, es, mi, il);
    A[8]  = pval((wv >> (16 + koff + 0)) & 1u, e2.x, es, mi, il);
    A[9]  = pval((wv >> (16 + koff + 1)) & 1u, e2.y, es, mi, il);
    A[10] = pval((wv >> (16 + koff + 2)) & 1u, e2.z, es, mi, il);
    A[11] = pval((wv >> (16 + koff + 3)) & 1u, e2.w, es, mi, il);
    A[12] = pval((wv >> (16 + koff + 4)) & 1u, e3.x, es, mi, il);
    A[13] = pval((wv >> (16 + koff + 5)) & 1u, e3.y, es, mi, il);
    A[14] = pval((wv >> (16 + koff + 6)) & 1u, e3.z, es, mi, il);
    A[15] = pval((wv >> (16 + koff + 7)) & 1u, e3.w, es, mi, il);

    const char* lbuf = smem + cur * LDS_BUF_;
#pragma unroll
    for (int ct = 0; ct < 4; ++ct) {
      const char* bp = lbuf + (ct * 16 + m16) * ROWB_ + jbo * 2;
      v8h blo = *(const v8h*)(bp);
      v8h bhi = *(const v8h*)(bp + 32);
      v16h Bv = __builtin_shufflevector(blo, bhi, 0, 1, 2, 3, 4, 5, 6, 7,
                                        8, 9, 10, 11, 12, 13, 14, 15);
      acc[ct] = wmma_f16(A, Bv, acc[ct]);
    }

    if ((wav == 0) && (c + 1 < N_ / 32))     // next chunk landed?
      __builtin_amdgcn_s_wait_tensorcnt(0);
    __syncthreads();                         // release buffers for next round
  }

  const int roff = hi ? 8 : 0;
#pragma unroll
  for (int ct = 0; ct < 4; ++ct) {
    const int d = ct * 16 + m16;
#pragma unroll
    for (int r = 0; r < 8; ++r) {
      const int row = i0 + r + roff;
      out[((size_t)(b * N_ + row)) * HD_ + h * D_ + d] = acc[ct][r];
    }
  }
}

extern "C" void kernel_launch(void* const* d_in, const int* in_sizes, int n_in,
                              void* d_out, int out_size, void* d_ws, size_t ws_size,
                              hipStream_t stream) {
  (void)in_sizes; (void)n_in; (void)out_size; (void)ws_size;
  const float* x       = (const float*)d_in[0];
  const int*   adj     = (const int*)d_in[1];
  const float* W       = (const float*)d_in[2];
  const float* att_src = (const float*)d_in[3];
  const float* att_dst = (const float*)d_in[4];
  float* out = (float*)d_out;

  // ws: hT f16 (4MB) | e_src | e_dst | m | 1/l (128KB each) | adjbits (512KB)
  _Float16* hT = (_Float16*)d_ws;
  float* e_src = (float*)((char*)d_ws + (size_t)B_ * H_ * D_ * N_ * sizeof(_Float16));
  float* e_dst = e_src + B_ * H_ * N_;
  float* mrow  = e_dst + B_ * H_ * N_;
  float* ilrow = mrow  + B_ * H_ * N_;
  unsigned* adjbits = (unsigned*)(ilrow + B_ * H_ * N_);

  k_linear_hT<<<256, 256, 0, stream>>>(x, W, hT);
  k_pack_adj<<<512, 256, 0, stream>>>(adj, adjbits);
  k_edge<<<128, 256, 0, stream>>>(hT, att_src, att_dst, e_src, e_dst);
  k_stats<<<4096, 256, 0, stream>>>(adjbits, e_src, e_dst, mrow, ilrow);
  k_attn<<<256, 256, 0, stream>>>(adjbits, hT, e_src, e_dst, mrow, ilrow, out);
}
